// Interact_89180700934376
// MI455X (gfx1250) — matrix operand order repaired and checked by reading, not compiled, not run on hardware
//
#include <hip/hip_runtime.h>
#include <cstdint>

#define BIGF 1.0e10f
#define NEG_INF (-3.0e38f)

typedef __attribute__((ext_vector_type(16))) __bf16   v16bf;
typedef __attribute__((ext_vector_type(8)))  float    v8f;
typedef __attribute__((ext_vector_type(4)))  unsigned int u32x4;
typedef __attribute__((ext_vector_type(4)))  float    f32x4;

typedef __attribute__((ext_vector_type(4))) unsigned int tdm_u4;
typedef __attribute__((ext_vector_type(8))) int          tdm_i8;
typedef __attribute__((ext_vector_type(4))) int          tdm_i4;

#if defined(__HIP_DEVICE_COMPILE__) && __has_builtin(__builtin_amdgcn_tensor_load_to_lds)
#define HAS_TDM 1
#else
#define HAS_TDM 0
#endif

union FragU { v16bf v; u32x4 u[2]; };

// f32 -> bf16 (round to nearest even)
__device__ __forceinline__ uint16_t f2bf(float x) {
    union { float f; unsigned int u; } c; c.f = x;
    unsigned int u = c.u;
    return (uint16_t)((u + 0x7FFFu + ((u >> 16) & 1u)) >> 16);
}

// A fragment (16x32 bf16) from LDS row-major [m][k]: lane=M, chunks at 8*half, 16+8*half
__device__ __forceinline__ v16bf load_a(const uint16_t* base, int rowBase, int stride, int k0, int lane) {
    int r = rowBase + (lane & 15);
    int h = (lane >> 4) & 1;
    const uint16_t* p = base + (size_t)r * stride + k0;
    FragU f;
    f.u[0] = *(const u32x4*)(p + 8 * h);
    f.u[1] = *(const u32x4*)(p + 16 + 8 * h);
    return f.v;
}

// A fragment from a per-lane global row pointer (row already selected by lane)
__device__ __forceinline__ v16bf load_a_g(const uint16_t* rowPtr, int k0, int h) {
    FragU f;
    f.u[0] = *(const u32x4*)(rowPtr + k0 + 8 * h);
    f.u[1] = *(const u32x4*)(rowPtr + k0 + 16 + 8 * h);
    return f.v;
}

// B fragment (32x16 bf16): lane=N col; chunks at 8*half, 16+8*half
// base stored transposed: base[n][k]
__device__ __forceinline__ v16bf load_b(const uint16_t* base, int colBase, int stride, int k0, int lane) {
    int c = colBase + (lane & 15);
    int h = (lane >> 4) & 1;
    const uint16_t* p = base + (size_t)c * stride + k0 + 8 * h;
    FragU f;
    f.u[0] = *(const u32x4*)(p);
    f.u[1] = *(const u32x4*)(p + 16);
    return f.v;
}

__device__ __forceinline__ v8f wmma_bf16(v16bf a, v16bf b, v8f c) {
    return __builtin_amdgcn_wmma_f32_16x16x32_bf16(false, a, false, b, (short)0, c, false, false);
}

#if HAS_TDM
// Issue a 2-D TDM load: tile (tile_d0 x tile_d1) of 8-byte elements from global
// (row stride d0_stride, in 8B units) into LDS at lds_addr (packed rows).
__device__ __forceinline__ void tdm_load_2d(unsigned int lds_addr, const void* gptr,
                                            unsigned int tensor_d0, unsigned int tensor_d1,
                                            unsigned int tile_d0, unsigned int tile_d1,
                                            unsigned long long d0_stride) {
    unsigned long long ga = (unsigned long long)(uintptr_t)gptr;
    tdm_u4 g0;
    g0[0] = 1u;                                             // count=1, user mode
    g0[1] = lds_addr;                                       // LDS byte address
    g0[2] = (unsigned int)(ga & 0xffffffffu);               // global addr lo
    g0[3] = (unsigned int)((ga >> 32) & 0x01ffffffu) | 0x80000000u;  // addr hi | type=2
    tdm_i8 g1;
    const unsigned int ds = 3u;                             // 8-byte elements
    g1[0] = (int)(ds << 16);                                // wg_mask=0 | data_size
    g1[1] = (int)((tensor_d0 & 0xffffu) << 16);             // abar=0 | tensor_dim0 lo
    g1[2] = (int)(((tensor_d0 >> 16) & 0xffffu) | ((tensor_d1 & 0xffffu) << 16));
    g1[3] = (int)(((tensor_d1 >> 16) & 0xffffu) | ((tile_d0 & 0xffffu) << 16));
    g1[4] = (int)(tile_d1 & 0xffffu);                       // tile_dim1 | tile_dim2=0
    g1[5] = (int)(unsigned int)(d0_stride & 0xffffffffu);   // dim0 stride lo
    g1[6] = (int)(unsigned int)((d0_stride >> 32) & 0xffffu); // stride hi | dim1 stride=0
    g1[7] = 0;
    tdm_i4 z4 = {0, 0, 0, 0};
#if __clang_major__ >= 23
    tdm_i8 z8 = {0, 0, 0, 0, 0, 0, 0, 0};
    __builtin_amdgcn_tensor_load_to_lds(g0, g1, z4, z4, z8, 0);
#else
    __builtin_amdgcn_tensor_load_to_lds(g0, g1, z4, z4, 0);
#endif
}

__device__ __forceinline__ unsigned int lds_off(const void* p) {
    return (unsigned int)(uintptr_t)p;   // low 32 bits of flat addr = LDS offset
}
#endif

// ---------------------------------------------------------------------------
// Kernel 0: v (f32) -> vbf [B][Lv][D] bf16  and  vtbf [B][D][Lv] bf16
// 64x64 tiles via LDS transpose; all global accesses coalesced.
// ---------------------------------------------------------------------------
__global__ __launch_bounds__(256) void v_convert(const float* __restrict__ v,
                                                 uint16_t* __restrict__ vbf,
                                                 uint16_t* __restrict__ vtbf,
                                                 int Lv, int D) {
    __shared__ uint16_t tile[64][66];
    const int tid = threadIdx.x;
    const int b  = blockIdx.z;
    const int l0 = blockIdx.x * 64;
    const int e0 = blockIdx.y * 64;
    for (int i = tid; i < 4096; i += 256) {
        int j = i >> 6, e = i & 63;
        tile[j][e] = f2bf(v[((size_t)b * Lv + l0 + j) * D + e0 + e]);
    }
    __syncthreads();
    for (int i = tid; i < 4096; i += 256) {
        int j = i >> 6, e = i & 63;
        vbf[((size_t)b * Lv + l0 + j) * D + e0 + e] = tile[j][e];
    }
    for (int i = tid; i < 4096; i += 256) {
        int e = i >> 6, j = i & 63;
        vtbf[((size_t)b * D + e0 + e) * Lv + l0 + j] = tile[j][e];
    }
}

// ---------------------------------------------------------------------------
// Kernel 1: qw = q @ W -> bf16 ws.  BM=64, BN=128, BK=32; 8 waves 4(M)x2(N).
// ---------------------------------------------------------------------------
__global__ __launch_bounds__(256) void qw_gemm(const float* __restrict__ q,
                                               const float* __restrict__ w,
                                               uint16_t* __restrict__ qw,
                                               int M, int D) {
    __shared__ __align__(16) uint16_t qt[64 * 32];    // A tile [m][k]
    __shared__ __align__(16) uint16_t kt[128 * 32];   // B tile transposed [n][k]

    const int tid  = threadIdx.x;
    const int lane = tid & 31;
    const int wave = tid >> 5;
    const int slab = wave & 3;
    const int grp  = wave >> 2;
    const int half = (lane >> 4) & 1;
    const int ln   = lane & 15;
    const int m0 = blockIdx.x * 64;
    const int n0 = blockIdx.y * 128;

    v8f acc[4];
    for (int nt = 0; nt < 4; ++nt)
        for (int g = 0; g < 8; ++g) acc[nt][g] = 0.0f;

    for (int k0 = 0; k0 < D; k0 += 32) {
        // stage q tile: 512 float4 loads, coalesced
        for (int i = tid; i < 512; i += 256) {
            int r = i >> 3, c4 = (i & 7) << 2;
            f32x4 x = *(const f32x4*)(q + (size_t)(m0 + r) * D + k0 + c4);
            int base = r * 32 + c4;
            qt[base + 0] = f2bf(x[0]); qt[base + 1] = f2bf(x[1]);
            qt[base + 2] = f2bf(x[2]); qt[base + 3] = f2bf(x[3]);
        }
        // stage W tile transposed: 1024 float4 loads, coalesced over n
        for (int i = tid; i < 1024; i += 256) {
            int c = i >> 5, n4 = (i & 31) << 2;
            f32x4 x = *(const f32x4*)(w + (size_t)(k0 + c) * 512 + n0 + n4);
            kt[(n4 + 0) * 32 + c] = f2bf(x[0]);
            kt[(n4 + 1) * 32 + c] = f2bf(x[1]);
            kt[(n4 + 2) * 32 + c] = f2bf(x[2]);
            kt[(n4 + 3) * 32 + c] = f2bf(x[3]);
        }
        __syncthreads();
        v16bf a = load_a(qt, slab * 16, 32, 0, lane);
        for (int nt = 0; nt < 4; ++nt) {
            v16bf b = load_b(kt, grp * 64 + nt * 16, 32, 0, lane);
            acc[nt] = wmma_bf16(a, b, acc[nt]);
        }
        __syncthreads();
    }
    for (int nt = 0; nt < 4; ++nt)
        for (int g = 0; g < 8; ++g) {
            int row = m0 + slab * 16 + half * 8 + g;
            int col = n0 + grp * 64 + nt * 16 + ln;
            qw[(size_t)row * 512 + col] = f2bf(acc[nt][g]);
        }
}

// ---------------------------------------------------------------------------
// Kernel 2: mv[b][e] = max_j ( v[b][j][e] - (1-mask[b][j])*BIG )
// ---------------------------------------------------------------------------
__global__ __launch_bounds__(128) void mv_reduce(const float* __restrict__ v,
                                                 const float* __restrict__ vmask,
                                                 float* __restrict__ mv,
                                                 int Lv, int D) {
    const int b = blockIdx.y;
    const int e = blockIdx.x * 128 + threadIdx.x;
    float m = NEG_INF;
    for (int j = 0; j < Lv; ++j) {
        float x = v[((size_t)b * Lv + j) * D + e] - (1.0f - vmask[(size_t)b * Lv + j]) * BIGF;
        m = fmaxf(m, x);
    }
    mv[b * D + e] = m;
}

// ---------------------------------------------------------------------------
// Kernel 3: flash attention. BM=64 queries/WG, BN=64 per Lv step.
// TDM double-buffers both v-tile copies; A frags stream from global (L2).
// ---------------------------------------------------------------------------
__global__ __launch_bounds__(256) void attn_kernel(const uint16_t* __restrict__ vbf,
                                                   const uint16_t* __restrict__ vtbf,
                                                   const float* __restrict__ vmask,
                                                   const uint16_t* __restrict__ qw,
                                                   float* __restrict__ out,
                                                   int Lq, int Lv, int D) {
    __shared__ __align__(16) uint16_t vrow[2][64 * 512];   // 2 x 64KB  S-phase B: v[j][e]
    __shared__ __align__(16) uint16_t vcol[2][512 * 64];   // 2 x 64KB  PV-phase B: v^T[e][j]
    __shared__ __align__(16) uint16_t pLds[64 * 64];       // 8KB       P tile [m][j]
    __shared__ float statsMax[64 * 2];
    __shared__ float statsSum[64 * 2];

    const int tid  = threadIdx.x;
    const int lane = tid & 31;
    const int wave = tid >> 5;
    const int slab = wave & 3;    // 16 M rows
    const int grp  = wave >> 2;   // S col half / O col half
    const int half = (lane >> 4) & 1;
    const int ln   = lane & 15;

    const int b  = blockIdx.y;
    const int m0 = blockIdx.x * 64;
    const int n0 = grp * 32;
    const int d0 = grp * 256;
    const int NT = Lv / 64;

    // per-lane A row pointer (qw bf16, row = this lane's M row)
    const uint16_t* qwRow = qw + ((size_t)b * Lq + m0 + slab * 16 + ln) * 512;

#if HAS_TDM
    // vrow tile t: contiguous 64KB; vcol tile t: 512 rows x 128B, row stride Lv*2 B
    #define ISSUE_TILE(t)                                                                 \
        do {                                                                              \
            int _l0 = (t) * 64;                                                           \
            tdm_load_2d(lds_off(vrow[(t) & 1]), vbf + ((size_t)b * Lv + _l0) * 512,       \
                        8192u, 1u, 8192u, 1u, 8192ull);                                   \
            tdm_load_2d(lds_off(vcol[(t) & 1]), vtbf + (size_t)b * 512 * Lv + _l0,        \
                        (unsigned)(Lv / 4), 512u, 16u, 512u, (unsigned long long)(Lv / 4)); \
        } while (0)
    if (wave == 0) ISSUE_TILE(0);
#endif

    v8f acc[16];
    for (int nt = 0; nt < 16; ++nt)
        for (int g = 0; g < 8; ++g) acc[nt][g] = 0.0f;
    float run_max[8], run_sum[8];
    for (int g = 0; g < 8; ++g) { run_max[g] = NEG_INF; run_sum[g] = 0.0f; }

    for (int t = 0; t < NT; ++t) {
        const int l0 = t * 64;
#if HAS_TDM
        if (wave == 0) {
            if (t + 1 < NT) {
                ISSUE_TILE(t + 1);
                __builtin_amdgcn_s_wait_tensorcnt(2);   // tile t's two loads complete
            } else {
                __builtin_amdgcn_s_wait_tensorcnt(0);   // final tile fully resident
            }
        }
#else
        // fallback: cooperative staging of tile t
        {
            const uint32_t* s0 = (const uint32_t*)(vbf + ((size_t)b * Lv + l0) * 512);
            uint32_t* dst0 = (uint32_t*)vrow[t & 1];
            for (int i = tid; i < 64 * 256; i += 256) dst0[i] = s0[i];
            for (int i = tid; i < 512 * 32; i += 256) {
                int e = i >> 5, j4 = (i & 31) << 1;
                const uint32_t* s1 = (const uint32_t*)(vtbf + ((size_t)b * 512 + e) * Lv + l0);
                ((uint32_t*)vcol[t & 1])[e * 32 + (j4 >> 1)] = s1[j4 >> 1];
            }
        }
#endif
        __syncthreads();   // tile t resident
        const uint16_t* vr = vrow[t & 1];
        const uint16_t* vc = vcol[t & 1];

        // ---- S = qw @ v^T : per wave 16x32 slice, K = 512 ----
        v8f s0, s1;
        for (int g = 0; g < 8; ++g) { s0[g] = 0.0f; s1[g] = 0.0f; }
        for (int kk = 0; kk < 512; kk += 32) {
            v16bf a  = load_a_g(qwRow, kk, half);
            v16bf b0 = load_b(vr, n0,      512, kk, lane);
            v16bf b1 = load_b(vr, n0 + 16, 512, kk, lane);
            s0 = wmma_bf16(a, b0, s0);
            s1 = wmma_bf16(a, b1, s1);
        }

        // ---- scale + additive mask ----
        float bias0 = (1.0f - vmask[(size_t)b * Lv + l0 + n0 + ln]) * BIGF;
        float bias1 = (1.0f - vmask[(size_t)b * Lv + l0 + n0 + 16 + ln]) * BIGF;
        float sv0[8], sv1[8], tmax[8];
        for (int g = 0; g < 8; ++g) {
            sv0[g] = s0[g] * 0.1f - bias0;
            sv1[g] = s1[g] * 0.1f - bias1;
            tmax[g] = fmaxf(sv0[g], sv1[g]);
        }
        for (int msk = 1; msk < 16; msk <<= 1)
            for (int g = 0; g < 8; ++g)
                tmax[g] = fmaxf(tmax[g], __shfl_xor(tmax[g], msk, 32));
        if (ln == 0)
            for (int g = 0; g < 8; ++g)
                statsMax[(slab * 16 + half * 8 + g) * 2 + grp] = tmax[g];
        __syncthreads();

        float mnew[8], alpha[8];
        for (int g = 0; g < 8; ++g) {
            int row = slab * 16 + half * 8 + g;
            float mt = fmaxf(statsMax[row * 2 + 0], statsMax[row * 2 + 1]);
            mnew[g]  = fmaxf(run_max[g], mt);
            alpha[g] = __expf(run_max[g] - mnew[g]);
            run_max[g] = mnew[g];
        }

        // ---- P = exp(S - mnew); partial sums; P -> LDS (bf16) ----
        float tsum[8];
        for (int g = 0; g < 8; ++g) {
            float p0 = __expf(sv0[g] - mnew[g]);
            float p1 = __expf(sv1[g] - mnew[g]);
            tsum[g] = p0 + p1;
            int row = slab * 16 + half * 8 + g;
            pLds[row * 64 + n0 + ln]      = f2bf(p0);
            pLds[row * 64 + n0 + 16 + ln] = f2bf(p1);
        }
        for (int msk = 1; msk < 16; msk <<= 1)
            for (int g = 0; g < 8; ++g)
                tsum[g] += __shfl_xor(tsum[g], msk, 32);
        if (ln == 0)
            for (int g = 0; g < 8; ++g)
                statsSum[(slab * 16 + half * 8 + g) * 2 + grp] = tsum[g];
        __syncthreads();

        for (int g = 0; g < 8; ++g) {
            int row = slab * 16 + half * 8 + g;
            run_sum[g] = run_sum[g] * alpha[g] + statsSum[row * 2 + 0] + statsSum[row * 2 + 1];
        }

        // ---- rescale O; O += P @ V (wave: 16 M rows x 256 Dout cols) ----
        for (int nt = 0; nt < 16; ++nt)
            for (int g = 0; g < 8; ++g) acc[nt][g] *= alpha[g];

        for (int kk = 0; kk < 64; kk += 32) {
            v16bf a = load_a(pLds, slab * 16, 64, kk, lane);
            for (int nt = 0; nt < 16; ++nt) {
                v16bf bb = load_b(vc, d0 + nt * 16, 64, kk, lane);
                acc[nt] = wmma_bf16(a, bb, acc[nt]);
            }
        }
        __syncthreads();   // all reads of buf[t&1]/pLds done before it is re-targeted
    }

    // ---- epilogue: O / run_sum -> out[..., 0:512) ----
    for (int nt = 0; nt < 16; ++nt)
        for (int g = 0; g < 8; ++g) {
            int row = m0 + slab * 16 + half * 8 + g;
            int e   = d0 + nt * 16 + ln;
            out[((size_t)b * Lq + row) * 1536 + e] = acc[nt][g] / run_sum[g];
        }
}

// ---------------------------------------------------------------------------
// Kernel 4: out[..., 512:1024) = q ; out[..., 1024:1536) = mv broadcast
// ---------------------------------------------------------------------------
__global__ __launch_bounds__(256) void copy_bcast(const float* __restrict__ q,
                                                  const float* __restrict__ mv,
                                                  float* __restrict__ out,
                                                  int Lq, int D) {
    size_t idx = (size_t)blockIdx.x * 256 + threadIdx.x;
    size_t bl  = idx / D;
    int e = (int)(idx % D);
    int b = (int)(bl / Lq);
    out[bl * 1536 + 512 + e]  = q[idx];
    out[bl * 1536 + 1024 + e] = mv[b * D + e];
}

// ---------------------------------------------------------------------------
extern "C" void kernel_launch(void* const* d_in, const int* in_sizes, int n_in,
                              void* d_out, int out_size, void* d_ws, size_t ws_size,
                              hipStream_t stream) {
    const float* q  = (const float*)d_in[0];   // [8, 2048, 512]
    const float* v  = (const float*)d_in[1];   // [8, 2048, 512]
    const float* vm = (const float*)d_in[2];   // [8, 2048, 1]
    const float* w  = (const float*)d_in[3];   // [512, 512]
    float* out = (float*)d_out;                // [8, 2048, 1536]

    const int B = 8, Lq = 2048, Lv = 2048, D = 512;
    const int M = B * Lq;                      // 16384

    char* ws = (char*)d_ws;
    uint16_t* qw_ws = (uint16_t*)ws;                                  // 16 MB
    uint16_t* vbf   = (uint16_t*)(ws + (size_t)M * 512 * 2);          // 16 MB
    uint16_t* vtbf  = (uint16_t*)(ws + (size_t)M * 512 * 4);          // 16 MB
    float*    mv_ws = (float*)   (ws + (size_t)M * 512 * 6);          // 16 KB

    v_convert<<<dim3(Lv / 64, D / 64, B), 256, 0, stream>>>(v, vbf, vtbf, Lv, D);
    qw_gemm<<<dim3(M / 64, 512 / 128), 256, 0, stream>>>(q, w, qw_ws, M, D);
    mv_reduce<<<dim3(D / 128, B), 128, 0, stream>>>(v, vm, mv_ws, Lv, D);
    attn_kernel<<<dim3(Lq / 64, B), 256, 0, stream>>>(vbf, vtbf, vm, qw_ws, out, Lq, Lv, D);
    copy_bcast<<<(int)(((size_t)B * Lq * D) / 256), 256, 0, stream>>>(q, mv_ws, out, Lq, D);
}
